// MultiHeadAttentionWithRoPE_11871289606378
// MI455X (gfx1250) — matrix operand order repaired and checked by reading, not compiled, not run on hardware
//
#include <hip/hip_runtime.h>
#include <math.h>

#define B_  2
#define S_  2048
#define H_  1024
#define NH_ 16
#define HD_ 64
#define M_  (B_ * S_) // 4096

typedef __bf16 bf16;
typedef __attribute__((ext_vector_type(16))) bf16  v16bf;
typedef __attribute__((ext_vector_type(8)))  bf16  v8bf;
typedef __attribute__((ext_vector_type(8)))  float v8f;

__device__ __forceinline__ v16bf pack2(v8bf a, v8bf b) {
  v16bf r;
#pragma unroll
  for (int i = 0; i < 8; ++i) { r[i] = a[i]; r[i + 8] = b[i]; }
  return r;
}

__device__ __forceinline__ v8f wmma_bf16(v16bf a, v16bf b, v8f c) {
  return __builtin_amdgcn_wmma_f32_16x16x32_bf16(false, a, false, b, (short)0, c,
                                                 false, false);
}

// ---------------------------------------------------------------- convert
__global__ void cvt_f32_bf16(const float* __restrict__ in, bf16* __restrict__ out,
                             int n) {
  int i = blockIdx.x * blockDim.x + threadIdx.x;
  if (i < n) out[i] = (bf16)in[i];
}

// ---------------------------------------------------------------- GEMM (C = A * W^T + bias)
// A: (M x K) bf16 row-major.  W: (N x K) bf16 row-major (so B[k][n] = W[n][k],
// contiguous along k).  Block = 4 waves, block tile 128M x 64N; each wave owns
// a 32M x 64N strip: 2 A-fragments, each B-fragment feeds 2 WMMAs (2x the
// L2-path arithmetic intensity of a 16-row strip).
// mode 0: bf16 out row-major. mode 1: bf16 out (b,h,d,s) transposed (for V).
// mode 2: fp32 out row-major.
__global__ __launch_bounds__(128) void gemm_wmma(const bf16* __restrict__ A,
                                                 const bf16* __restrict__ W,
                                                 const float* __restrict__ bias,
                                                 void* __restrict__ outp, int N,
                                                 int K, int mode) {
  const int lane = threadIdx.x & 31;
  const int wave = threadIdx.x >> 5;
  const int lo = lane & 15, hi = lane >> 4;
  const int m0 = blockIdx.y * 128 + wave * 32;
  const int n0 = blockIdx.x * 64;

  v8f acc[2][4] = {};
  const bf16* arow0 = A + (size_t)(m0 + lo) * K;
  const bf16* arow1 = A + (size_t)(m0 + 16 + lo) * K;
  for (int k0 = 0; k0 < K; k0 += 32) {
    const int kb = k0 + hi * 16;
    const v16bf a0 =
        pack2(*(const v8bf*)(arow0 + kb), *(const v8bf*)(arow0 + kb + 8));
    const v16bf a1 =
        pack2(*(const v8bf*)(arow1 + kb), *(const v8bf*)(arow1 + kb + 8));
#pragma unroll
    for (int t = 0; t < 4; ++t) {
      const int col = n0 + t * 16 + lo;
      const v16bf bfrag = *(const v16bf*)(W + (size_t)col * K + k0 + hi * 16);
      acc[0][t] = wmma_bf16(a0, bfrag, acc[0][t]);
      acc[1][t] = wmma_bf16(a1, bfrag, acc[1][t]);
    }
  }
#pragma unroll
  for (int r = 0; r < 2; ++r) {
#pragma unroll
    for (int t = 0; t < 4; ++t) {
#pragma unroll
      for (int v = 0; v < 8; ++v) {
        const int mm = m0 + 16 * r + v + 8 * hi;   // C layout: row = v + 8*hi
        const int nn = n0 + t * 16 + lo;           // col = lane&15
        const float val = acc[r][t][v] + bias[nn];
        if (mode == 0) {
          ((bf16*)outp)[(size_t)mm * N + nn] = (bf16)val;
        } else if (mode == 1) {                    // V -> (b,h,d,s)
          const int b = mm >> 11, s = mm & (S_ - 1);
          const int h = nn >> 6, d = nn & (HD_ - 1);
          ((bf16*)outp)[(((size_t)b * NH_ + h) * HD_ + d) * S_ + s] = (bf16)val;
        } else {
          ((float*)outp)[(size_t)mm * N + nn] = val;
        }
      }
    }
  }
}

// ---------------------------------------------------------------- RoPE (in place, bf16)
__global__ void rope_kernel(bf16* __restrict__ Q, bf16* __restrict__ Kb) {
  const int idx = blockIdx.x * blockDim.x + threadIdx.x;
  if (idx >= B_ * S_ * NH_ * (HD_ / 2)) return;
  const int d = idx & 31;
  const int h = (idx >> 5) & (NH_ - 1);
  const int s = (idx >> 9) & (S_ - 1);
  const int b = idx >> 20;
  const float inv = __powf(10000.0f, -(float)(2 * d) * (1.0f / (float)HD_));
  const float th = (float)s * inv;
  float sn, cs;
  __sincosf(th, &sn, &cs);
  const size_t base = (((size_t)b * S_ + s) * NH_ + h) * HD_;
  {
    const float x1 = (float)Q[base + d], x2 = (float)Q[base + d + 32];
    Q[base + d] = (bf16)(x1 * cs - x2 * sn);
    Q[base + d + 32] = (bf16)(x2 * cs + x1 * sn);
  }
  {
    const float x1 = (float)Kb[base + d], x2 = (float)Kb[base + d + 32];
    Kb[base + d] = (bf16)(x1 * cs - x2 * sn);
    Kb[base + d + 32] = (bf16)(x2 * cs + x1 * sn);
  }
}

// ---------------------------------------------------------------- flash attention
// Q,K: (b,s,h,d) bf16 (post-RoPE). VT: (b,h,d,s) bf16. O: (b,s,h,d) bf16.
// Block = 4 waves; wave w owns q-rows [blk*64 + 16w, +16). Block-uniform key loop.
__global__ __launch_bounds__(128) void attn_kernel(const bf16* __restrict__ Qb,
                                                   const bf16* __restrict__ Kb,
                                                   const bf16* __restrict__ VT,
                                                   bf16* __restrict__ O) {
  __shared__ __attribute__((aligned(32))) bf16 pLds[4][16][48]; // 48 = pad to 96B rows
  const int lane = threadIdx.x & 31;
  const int wave = threadIdx.x >> 5;
  const int lo = lane & 15, hi = lane >> 4;
  const int b = blockIdx.z, h = blockIdx.y;
  const int qr0 = blockIdx.x * 64 + wave * 16;
  const int qend = blockIdx.x * 64 + 64; // uniform across block (causal bound)

  const bf16* qrow = Qb + (((size_t)b * S_ + qr0 + lo) * NH_ + h) * HD_;
  const int kb = hi * 16;
  const v16bf aq0 =
      pack2(*(const v8bf*)(qrow + kb), *(const v8bf*)(qrow + kb + 8));
  const v16bf aq1 =
      pack2(*(const v8bf*)(qrow + 32 + kb), *(const v8bf*)(qrow + 32 + kb + 8));

  v8f o0 = {}, o1 = {}, o2 = {}, o3 = {};
  float mrow[8], lrow[8];
#pragma unroll
  for (int v = 0; v < 8; ++v) { mrow[v] = -1e30f; lrow[v] = 0.0f; }

  for (int j = 0; j < qend; j += 32) {
    v8f s0 = {}, s1 = {};
    { // scores: Q(16x64) . K^T(64x32keys), two key tiles
      const bf16* k0p = Kb + (((size_t)b * S_ + j + lo) * NH_ + h) * HD_ + hi * 16;
      const bf16* k1p =
          Kb + (((size_t)b * S_ + j + 16 + lo) * NH_ + h) * HD_ + hi * 16;
      s0 = wmma_bf16(aq0, *(const v16bf*)(k0p), s0);
      s0 = wmma_bf16(aq1, *(const v16bf*)(k0p + 32), s0);
      s1 = wmma_bf16(aq0, *(const v16bf*)(k1p), s1);
      s1 = wmma_bf16(aq1, *(const v16bf*)(k1p + 32), s1);
    }
    const float scale = 0.125f; // 1/sqrt(64)
#pragma unroll
    for (int v = 0; v < 8; ++v) {
      const int qpos = qr0 + v + 8 * hi;
      const float x0 = (j + lo <= qpos) ? s0[v] * scale : -1e30f;
      const float x1 = (j + 16 + lo <= qpos) ? s1[v] * scale : -1e30f;
      float mt = fmaxf(x0, x1);
#pragma unroll
      for (int off = 1; off < 16; off <<= 1)
        mt = fmaxf(mt, __shfl_xor(mt, off, 32)); // xor bits 0..3 stay in 16-lane group
      const float mnew = fmaxf(mrow[v], mt);
      const float alpha = __expf(mrow[v] - mnew);
      const float p0 = __expf(x0 - mnew);
      const float p1 = __expf(x1 - mnew);
      float rs = p0 + p1;
#pragma unroll
      for (int off = 1; off < 16; off <<= 1) rs += __shfl_xor(rs, off, 32);
      lrow[v] = lrow[v] * alpha + rs;
      mrow[v] = mnew;
      o0[v] *= alpha; o1[v] *= alpha; o2[v] *= alpha; o3[v] *= alpha;
      pLds[wave][v + 8 * hi][lo] = (bf16)p0;       // C-layout -> row-major tile
      pLds[wave][v + 8 * hi][lo + 16] = (bf16)p1;
    }
    __syncthreads();
    // re-load P as an A-fragment (16 x 32 keys)
    const v16bf pA = pack2(*(const v8bf*)(&pLds[wave][lo][kb]),
                           *(const v8bf*)(&pLds[wave][lo][kb + 8]));
    { // O += P . V   (V^T layout => 16 contiguous keys per lane)
      const bf16* vbase =
          VT + ((size_t)b * NH_ + h) * HD_ * S_ + j + hi * 16;
      o0 = wmma_bf16(pA, *(const v16bf*)(vbase + (size_t)(lo)*S_), o0);
      o1 = wmma_bf16(pA, *(const v16bf*)(vbase + (size_t)(16 + lo) * S_), o1);
      o2 = wmma_bf16(pA, *(const v16bf*)(vbase + (size_t)(32 + lo) * S_), o2);
      o3 = wmma_bf16(pA, *(const v16bf*)(vbase + (size_t)(48 + lo) * S_), o3);
    }
    __syncthreads();
  }

#pragma unroll
  for (int v = 0; v < 8; ++v) {
    const float inv_l = 1.0f / lrow[v];
    const int s = qr0 + v + 8 * hi;
    bf16* orow = O + (((size_t)b * S_ + s) * NH_ + h) * HD_;
    orow[lo] = (bf16)(o0[v] * inv_l);
    orow[16 + lo] = (bf16)(o1[v] * inv_l);
    orow[32 + lo] = (bf16)(o2[v] * inv_l);
    orow[48 + lo] = (bf16)(o3[v] * inv_l);
  }
}

// ---------------------------------------------------------------- launch
extern "C" void kernel_launch(void* const* d_in, const int* in_sizes, int n_in,
                              void* d_out, int out_size, void* d_ws,
                              size_t ws_size, hipStream_t stream) {
  (void)in_sizes; (void)n_in; (void)out_size;
  if (ws_size < (size_t)(48u << 20)) return; // need 48 MB scratch

  const float* x  = (const float*)d_in[0];
  const float* Wq = (const float*)d_in[1];
  const float* bq = (const float*)d_in[2];
  const float* Wk = (const float*)d_in[3];
  const float* bk = (const float*)d_in[4];
  const float* Wv = (const float*)d_in[5];
  const float* bv = (const float*)d_in[6];
  const float* Wo = (const float*)d_in[7];
  const float* bo = (const float*)d_in[8];
  // d_in[9] = causal mask (implemented analytically)

  char* ws = (char*)d_ws;
  bf16* xb  = (bf16*)(ws);                 // 8 MB
  bf16* Wqb = (bf16*)(ws + (8u << 20));    // 2 MB each
  bf16* Wkb = (bf16*)(ws + (10u << 20));
  bf16* Wvb = (bf16*)(ws + (12u << 20));
  bf16* Wob = (bf16*)(ws + (14u << 20));
  bf16* Qb  = (bf16*)(ws + (16u << 20));   // 8 MB each
  bf16* Kbp = (bf16*)(ws + (24u << 20));
  bf16* VT  = (bf16*)(ws + (32u << 20));
  bf16* Ab  = (bf16*)(ws + (40u << 20));

  const int nx = M_ * H_, nw = H_ * H_;
  cvt_f32_bf16<<<(nx + 255) / 256, 256, 0, stream>>>(x, xb, nx);
  cvt_f32_bf16<<<(nw + 255) / 256, 256, 0, stream>>>(Wq, Wqb, nw);
  cvt_f32_bf16<<<(nw + 255) / 256, 256, 0, stream>>>(Wk, Wkb, nw);
  cvt_f32_bf16<<<(nw + 255) / 256, 256, 0, stream>>>(Wv, Wvb, nw);
  cvt_f32_bf16<<<(nw + 255) / 256, 256, 0, stream>>>(Wo, Wob, nw);

  dim3 gg(H_ / 64, M_ / 128); // 128-row block tiles (32 rows per wave)
  gemm_wmma<<<gg, 128, 0, stream>>>(xb, Wqb, bq, Qb, H_, H_, 0);
  gemm_wmma<<<gg, 128, 0, stream>>>(xb, Wkb, bk, Kbp, H_, H_, 0);
  gemm_wmma<<<gg, 128, 0, stream>>>(xb, Wvb, bv, VT, H_, H_, 1);

  const int nr = B_ * S_ * NH_ * (HD_ / 2);
  rope_kernel<<<(nr + 255) / 256, 256, 0, stream>>>(Qb, Kbp);

  attn_kernel<<<dim3(S_ / 64, NH_, B_), 128, 0, stream>>>(Qb, Kbp, VT, Ab);

  gemm_wmma<<<gg, 128, 0, stream>>>(Ab, Wob, bo, d_out, H_, H_, 2);
}
